// MyCNN_69801808495355
// MI455X (gfx1250) — compile-verified, matrix operand order checked by reading, not run on hardware
//
#include <hip/hip_runtime.h>
#include <hip/hip_bf16.h>
#include <math.h>

typedef __attribute__((ext_vector_type(16))) _Float16 v16h;
typedef __attribute__((ext_vector_type(8)))  _Float16 v8h;
typedef __attribute__((ext_vector_type(8)))  float    v8f;

// ---------------- workspace layout (bytes) ----------------
static constexpr size_t OFF_W2F16  = 0;                    // 9*64*40 f16 (tap-major, padded rows) = 46,080
static constexpr size_t OFF_FC1F16 = 46080;                // 128*3136 f16          = 802,816
static constexpr size_t OFF_H1     = 848896;               // 4096*32*196 f16       = 51,380,224
static constexpr size_t OFF_C2RAW  = 52229120;             // 4096*64*196 f32       = 205,520,896
static constexpr size_t OFF_H2     = 257750016;            // 4096*3136 f16         = 25,690,112
static constexpr size_t OFF_FEAT   = 283440128;            // 4096*128 f32          = 2,097,152
static constexpr size_t OFF_ACC    = 285537280;            // 192 f32 (sum1/sq1/sum2/sq2)
static constexpr size_t OFF_MEAN1  = 285538048;            // 32 f32
static constexpr size_t OFF_VAR1   = 285538176;            // 32 f32
static constexpr size_t OFF_MEAN2  = 285538304;            // 64 f32
static constexpr size_t OFF_VAR2   = 285538560;            // 64 f32
static constexpr size_t OFF_P      = 285538816;            // 129*129 f32           = 66,564
static constexpr size_t OFF_WOUT   = 285605504;            // 129*10 f32

// ---------------- prep: weight relayout/convert + zero accumulators ----------------
// conv2_w [co][ci][ky][kx] f32 -> wt [tap][co][ci(pad40)] f16 ; fc1_w -> f16 flat.
__global__ __launch_bounds__(256)
void prep_kernel(const float* __restrict__ c2w, const float* __restrict__ fw,
                 _Float16* __restrict__ wt, _Float16* __restrict__ fwh,
                 float* __restrict__ acc) {
    int i = blockIdx.x * 256 + threadIdx.x;
    if (i < 192) acc[i] = 0.f;
    if (i < 9 * 64 * 40) {
        int tapco = i / 40;
        int ci    = i - tapco * 40;
        int tap   = tapco >> 6;
        int co    = tapco & 63;
        wt[i] = (ci < 32) ? (_Float16)c2w[co * 288 + ci * 9 + tap] : (_Float16)0.f;
    }
    if (i < 401408) fwh[i] = (_Float16)fw[i];
}

// ---------------- conv1 batch statistics (recompute conv, reduce) ----------------
__global__ __launch_bounds__(256)
void conv1_stats(const float* __restrict__ x, const float* __restrict__ w1,
                 const float* __restrict__ cb, float* __restrict__ sum1, float* __restrict__ sq1) {
    int bc = blockIdx.x;                 // 4096*32 blocks
    int b = bc >> 5, co = bc & 31;
    __shared__ float xs[784];
    __shared__ float red[256], red2[256];
    for (int i = threadIdx.x; i < 784; i += 256) xs[i] = x[b * 784 + i];
    __syncthreads();
    float w[9];
#pragma unroll
    for (int t = 0; t < 9; ++t) w[t] = w1[co * 9 + t];
    float bias = cb[co];
    float s = 0.f, s2 = 0.f;
    for (int p = threadIdx.x; p < 784; p += 256) {
        int yy = p / 28, xx = p - yy * 28;
        float a = bias;
#pragma unroll
        for (int ky = 0; ky < 3; ++ky)
#pragma unroll
            for (int kx = 0; kx < 3; ++kx) {
                int iy = yy + ky - 1, ix = xx + kx - 1;
                if (iy >= 0 && iy < 28 && ix >= 0 && ix < 28)
                    a = fmaf(w[ky * 3 + kx], xs[iy * 28 + ix], a);
            }
        s += a; s2 = fmaf(a, a, s2);
    }
    red[threadIdx.x] = s; red2[threadIdx.x] = s2;
    __syncthreads();
    for (int off = 128; off > 0; off >>= 1) {
        if (threadIdx.x < off) {
            red[threadIdx.x]  += red[threadIdx.x + off];
            red2[threadIdx.x] += red2[threadIdx.x + off];
        }
        __syncthreads();
    }
    if (threadIdx.x == 0) { atomicAdd(&sum1[co], red[0]); atomicAdd(&sq1[co], red2[0]); }
}

__global__ void bn_finalize(const float* __restrict__ sum, const float* __restrict__ sq,
                            float* __restrict__ mean, float* __restrict__ var, int C, float invN) {
    int c = threadIdx.x;
    if (c < C) { float m = sum[c] * invN; mean[c] = m; var[c] = sq[c] * invN - m * m; }
}

// ---------------- conv1 + BN1 + ReLU + maxpool2 -> h1 (f16 NCHW) ----------------
__global__ __launch_bounds__(256)
void conv1_bn_pool(const float* __restrict__ x, const float* __restrict__ w1,
                   const float* __restrict__ cb, const float* __restrict__ mean,
                   const float* __restrict__ var, const float* __restrict__ g,
                   const float* __restrict__ bb, _Float16* __restrict__ h1) {
    int bc = blockIdx.x;
    int b = bc >> 5, co = bc & 31;
    __shared__ float xs[784];
    __shared__ float cv[784];
    for (int i = threadIdx.x; i < 784; i += 256) xs[i] = x[b * 784 + i];
    __syncthreads();
    float w[9];
#pragma unroll
    for (int t = 0; t < 9; ++t) w[t] = w1[co * 9 + t];
    float bias  = cb[co];
    float scale = g[co] * rsqrtf(var[co] + 1e-5f);
    float shift = bb[co] - mean[co] * scale;
    for (int p = threadIdx.x; p < 784; p += 256) {
        int yy = p / 28, xx = p - yy * 28;
        float a = bias;
#pragma unroll
        for (int ky = 0; ky < 3; ++ky)
#pragma unroll
            for (int kx = 0; kx < 3; ++kx) {
                int iy = yy + ky - 1, ix = xx + kx - 1;
                if (iy >= 0 && iy < 28 && ix >= 0 && ix < 28)
                    a = fmaf(w[ky * 3 + kx], xs[iy * 28 + ix], a);
            }
        cv[p] = fmaxf(fmaf(a, scale, shift), 0.f);
    }
    __syncthreads();
    for (int p = threadIdx.x; p < 196; p += 256) {
        int py = p / 14, px = p - py * 14;
        int y0 = py * 2, x0 = px * 2;
        float m = fmaxf(fmaxf(cv[y0 * 28 + x0], cv[y0 * 28 + x0 + 1]),
                        fmaxf(cv[(y0 + 1) * 28 + x0], cv[(y0 + 1) * 28 + x0 + 1]));
        h1[(size_t)(b * 32 + co) * 196 + p] = (_Float16)m;
    }
}

// ---------------- conv2 as 9-tap GEMM decomposition with WMMA ----------------
// Per tap: out(64xN) += W_tap(64x32) @ h1_shift(32xN); each tile-tap = ONE 16x16x32 WMMA.
// One block per batch image. h1 staged in LDS transposed [s][ci] with rows padded to 40
// halves (80B -> distinct bank groups for 16 consecutive n) plus ONE ZERO ROW (row 196):
// out-of-bounds taps redirect the LDS address to the zero row (single v_cndmask on the
// 32-bit LDS address, no data masking, no EXEC manipulation). Tap shifts are the
// compile-time byte offsets ((ky-1)*14+(kx-1))*40 halves off a per-tile base.
__global__ __launch_bounds__(256)
void conv2_wmma(const _Float16* __restrict__ wt, const _Float16* __restrict__ h1,
                const float* __restrict__ c2b, float* __restrict__ c2raw) {
    __shared__ alignas(16) _Float16 wl[9 * 64 * 40];   // 46,080 B
    __shared__ alignas(16) _Float16 hs[197 * 40];      // 15,760 B (row 196 = zeros)
    int b = blockIdx.x;
    for (int i = threadIdx.x; i < 9 * 64 * 40; i += 256) wl[i] = wt[i];
    for (int i = threadIdx.x; i < 32 * 196; i += 256) {
        int ci = i / 196, s = i - ci * 196;
        hs[s * 40 + ci] = h1[(size_t)(b * 32 + ci) * 196 + s];
    }
    for (int i = threadIdx.x; i < 40; i += 256) hs[196 * 40 + i] = (_Float16)0.f;
    __syncthreads();

    int wave = threadIdx.x >> 5;
    int lane = threadIdx.x & 31;
    int nl = lane & 15;
    int hh = lane >> 4;                  // half selector per ISA 16-bit fragment layout

    // 13 n-tiles (196 -> pad 208) x 4 m-tiles = 52 wave-tiles, 8 waves round-robin
    for (int tile = wave; tile < 52; tile += 8) {
        int m_tile = tile & 3;
        int n_tile = tile >> 2;
        int n = n_tile * 16 + nl;        // 0..207
        bool vn = n < 196;
        int nc = vn ? n : 0;
        int yy = nc / 14;
        int xx = nc - yy * 14;
        // per-tile boundary predicates (tap validity = vn && my[ky] && mx[kx])
        bool my0 = yy >= 1, my2 = yy <= 12;
        bool mx0 = xx >= 1, mx2 = xx <= 12;
        const _Float16* base = &hs[nc * 40 + hh * 8];
        const _Float16* zrow = &hs[196 * 40 + hh * 8];
        v8f acc = {};
#pragma unroll
        for (int t9 = 0; t9 < 9; ++t9) {
            const int ky = t9 / 3, kx = t9 - (t9 / 3) * 3;
            const _Float16* arow = &wl[(t9 * 64 + m_tile * 16 + nl) * 40 + hh * 8];
            v8h alo = *(const v8h*)(arow);
            v8h ahi = *(const v8h*)(arow + 16);
            bool ok = vn && (ky == 0 ? my0 : (ky == 2 ? my2 : true))
                         && (kx == 0 ? mx0 : (kx == 2 ? mx2 : true));
            const _Float16* brow = ok ? (base + ((ky - 1) * 14 + (kx - 1)) * 40) : zrow;
            v8h blo = *(const v8h*)(brow);
            v8h bhi = *(const v8h*)(brow + 16);
            v16h a, bf;
#pragma unroll
            for (int t = 0; t < 8; ++t) {
                a[t] = alo[t];  a[t + 8] = ahi[t];
                bf[t] = blo[t]; bf[t + 8] = bhi[t];
            }
            acc = __builtin_amdgcn_wmma_f32_16x16x32_f16(false, a, false, bf,
                                                         (short)0, acc, false, false);
        }
        if (vn) {
#pragma unroll
            for (int r = 0; r < 8; ++r) {
                int co = m_tile * 16 + r + 8 * hh;      // C layout: lane<16 M=r, lane>=16 M=r+8
                c2raw[(size_t)(b * 64 + co) * 196 + n] = acc[r] + c2b[co];
            }
        }
    }
}

// ---------------- BN2 statistics over conv2 raw output ----------------
__global__ __launch_bounds__(256)
void bn2_stats(const float* __restrict__ c2raw, float* __restrict__ sum2, float* __restrict__ sq2) {
    int co = blockIdx.x;                 // 64
    int chunk = blockIdx.y;              // 512 chunks of 8 batch images
    __shared__ float red[256], red2[256];
    float s = 0.f, s2 = 0.f;
    for (int i = threadIdx.x; i < 8 * 196; i += 256) {
        int b  = chunk * 8 + i / 196;
        int sp = i % 196;
        float v = c2raw[(size_t)(b * 64 + co) * 196 + sp];
        s += v; s2 = fmaf(v, v, s2);
    }
    red[threadIdx.x] = s; red2[threadIdx.x] = s2;
    __syncthreads();
    for (int off = 128; off > 0; off >>= 1) {
        if (threadIdx.x < off) {
            red[threadIdx.x]  += red[threadIdx.x + off];
            red2[threadIdx.x] += red2[threadIdx.x + off];
        }
        __syncthreads();
    }
    if (threadIdx.x == 0) { atomicAdd(&sum2[co], red[0]); atomicAdd(&sq2[co], red2[0]); }
}

// ---------------- BN2 + ReLU + maxpool2 -> h2 (f16 [B,3136]) ----------------
__global__ __launch_bounds__(256)
void bn2_pool(const float* __restrict__ c2raw, const float* __restrict__ mean,
              const float* __restrict__ var, const float* __restrict__ g,
              const float* __restrict__ bb, _Float16* __restrict__ h2) {
    int i = blockIdx.x * 256 + threadIdx.x;
    if (i >= 4096 * 64 * 49) return;
    int p  = i % 49;
    int t  = i / 49;
    int co = t % 64;
    int b  = t / 64;
    float scale = g[co] * rsqrtf(var[co] + 1e-5f);
    float shift = bb[co] - mean[co] * scale;
    int py = p / 7, px = p - py * 7;
    const float* base = &c2raw[(size_t)(b * 64 + co) * 196];
    float m = -INFINITY;
#pragma unroll
    for (int ky = 0; ky < 2; ++ky)
#pragma unroll
        for (int kx = 0; kx < 2; ++kx) {
            float v = fmaxf(fmaf(base[(py * 2 + ky) * 14 + px * 2 + kx], scale, shift), 0.f);
            m = fmaxf(m, v);
        }
    h2[(size_t)b * 3136 + co * 49 + p] = (_Float16)m;
}

// ---------------- fc1 GEMM with WMMA: feature = relu(h2 @ fc1_w^T + b) ----------------
__global__ __launch_bounds__(256)
void fc1_wmma(const _Float16* __restrict__ fwh, const _Float16* __restrict__ h2,
              const float* __restrict__ fb, float* __restrict__ feat) {
    int wave = threadIdx.x >> 5;
    int lane = threadIdx.x & 31;
    int m_tile = wave;                   // 8 M-tiles == 8 waves
    int n_tile = blockIdx.x;             // 256 N-tiles
    int nl = lane & 15;
    int hh = lane >> 4;

    const _Float16* arow = fwh + (size_t)(m_tile * 16 + nl) * 3136 + hh * 8;
    const _Float16* brow = h2  + (size_t)(n_tile * 16 + nl) * 3136 + hh * 8;

    v8f acc = {};
    for (int kb = 0; kb < 3136; kb += 32) {
        v8h alo = *(const v8h*)(arow + kb);
        v8h ahi = *(const v8h*)(arow + kb + 16);
        v8h blo = *(const v8h*)(brow + kb);
        v8h bhi = *(const v8h*)(brow + kb + 16);
        v16h a, bf;
#pragma unroll
        for (int t = 0; t < 8; ++t) {
            a[t] = alo[t];  a[t + 8] = ahi[t];
            bf[t] = blo[t]; bf[t + 8] = bhi[t];
        }
        acc = __builtin_amdgcn_wmma_f32_16x16x32_f16(false, a, false, bf,
                                                     (short)0, acc, false, false);
    }
#pragma unroll
    for (int r = 0; r < 8; ++r) {
        int j = m_tile * 16 + r + 8 * hh;
        int bidx = n_tile * 16 + nl;
        feat[(size_t)bidx * 128 + j] = fmaxf(acc[r] + fb[j], 0.f);
    }
}

// ---------------- sequential RLS scan (one workgroup; P symmetric) ----------------
__global__ __launch_bounds__(256)
void rls_kernel(const float* __restrict__ feat, const float* __restrict__ yv,
                const float* __restrict__ W0, const float* __restrict__ P0,
                const int* __restrict__ sep, float* __restrict__ P, float* __restrict__ Wout) {
    __shared__ float xs[129], Pxs[129], errs[10], Wl[1290];
    __shared__ float invden;
    int tid = threadIdx.x;
    for (int i = tid; i < 1290; i += 256) Wl[i] = W0[i];
    int dosep = *sep;
    if (dosep) {
        for (int i = tid; i < 16641; i += 256) P[i] = P0[i];
        __syncthreads();
        for (int it = 0; it < 4096; ++it) {
            if (tid < 129) xs[tid] = (tid < 128) ? feat[it * 128 + tid] : 1.f;
            __syncthreads();
            if (tid < 129) {                       // Px = P @ x
                const float* Pr = &P[tid * 129];
                float s = 0.f;
                for (int j = 0; j < 129; ++j) s = fmaf(Pr[j], xs[j], s);
                Pxs[tid] = s;
            }
            __syncthreads();
            if (tid == 0) {                        // denom = 1 + x.Px
                float d = 1.f;
                for (int j = 0; j < 129; ++j) d = fmaf(xs[j], Pxs[j], d);
                invden = 1.f / d;
            }
            if (tid >= 32 && tid < 42) {           // err = y - x@W
                int c = tid - 32;
                float e = yv[it * 10 + c];
                for (int j = 0; j < 129; ++j) e = fmaf(-xs[j], Wl[j * 10 + c], e);
                errs[c] = e;
            }
            __syncthreads();
            float inv = invden;
            for (int i = tid; i < 1290; i += 256) { // W += outer(Px, err)/denom
                int r = i / 10, c = i - r * 10;
                Wl[i] = fmaf(Pxs[r] * errs[c], inv, Wl[i]);
            }
            for (int i = tid; i < 16641; i += 256) { // P -= outer(Px, Px)/denom (symmetric)
                int r = i / 129, c = i - r * 129;
                P[i] = fmaf(-Pxs[r] * Pxs[c], inv, P[i]);
            }
            __syncthreads();
        }
    }
    for (int i = tid; i < 1290; i += 256) Wout[i] = Wl[i];
}

// ---------------- final apply: out = feature @ W[:-1] + W[-1] ----------------
__global__ __launch_bounds__(256)
void final_gemm(const float* __restrict__ feat, const float* __restrict__ Wout,
                float* __restrict__ out) {
    int i = blockIdx.x * 256 + threadIdx.x;
    if (i >= 4096 * 10) return;
    int c = i % 10, b = i / 10;
    float s = Wout[128 * 10 + c];
    const float* f = &feat[(size_t)b * 128];
    for (int j = 0; j < 128; ++j) s = fmaf(f[j], Wout[j * 10 + c], s);
    out[i] = s;
}

extern "C" void kernel_launch(void* const* d_in, const int* in_sizes, int n_in,
                              void* d_out, int out_size, void* d_ws, size_t ws_size,
                              hipStream_t stream) {
    const float* x   = (const float*)d_in[0];
    const float* yv  = (const float*)d_in[1];
    const float* c1w = (const float*)d_in[2];
    const float* c1b = (const float*)d_in[3];
    const float* g1  = (const float*)d_in[4];
    const float* b1  = (const float*)d_in[5];
    const float* c2w = (const float*)d_in[6];
    const float* c2b = (const float*)d_in[7];
    const float* g2  = (const float*)d_in[8];
    const float* b2  = (const float*)d_in[9];
    const float* fw  = (const float*)d_in[10];
    const float* fb  = (const float*)d_in[11];
    const float* W0  = (const float*)d_in[12];
    const float* P0  = (const float*)d_in[13];
    const int*   sep = (const int*)d_in[14];

    char* ws = (char*)d_ws;
    _Float16* wt    = (_Float16*)(ws + OFF_W2F16);
    _Float16* fwh   = (_Float16*)(ws + OFF_FC1F16);
    _Float16* h1    = (_Float16*)(ws + OFF_H1);
    float*    c2raw = (float*)(ws + OFF_C2RAW);
    _Float16* h2    = (_Float16*)(ws + OFF_H2);
    float*    feat  = (float*)(ws + OFF_FEAT);
    float*    acc   = (float*)(ws + OFF_ACC);
    float*    sum1  = acc + 0,  *sq1 = acc + 32;
    float*    sum2  = acc + 64, *sq2 = acc + 128;
    float*    mean1 = (float*)(ws + OFF_MEAN1);
    float*    var1  = (float*)(ws + OFF_VAR1);
    float*    mean2 = (float*)(ws + OFF_MEAN2);
    float*    var2  = (float*)(ws + OFF_VAR2);
    float*    Pws   = (float*)(ws + OFF_P);
    float*    Wout  = (float*)(ws + OFF_WOUT);

    prep_kernel<<<1568, 256, 0, stream>>>(c2w, fw, wt, fwh, acc);
    conv1_stats<<<4096 * 32, 256, 0, stream>>>(x, c1w, c1b, sum1, sq1);
    bn_finalize<<<1, 64, 0, stream>>>(sum1, sq1, mean1, var1, 32, 1.f / (4096.f * 784.f));
    conv1_bn_pool<<<4096 * 32, 256, 0, stream>>>(x, c1w, c1b, mean1, var1, g1, b1, h1);
    conv2_wmma<<<4096, 256, 0, stream>>>(wt, h1, c2b, c2raw);
    bn2_stats<<<dim3(64, 512), 256, 0, stream>>>(c2raw, sum2, sq2);
    bn_finalize<<<1, 64, 0, stream>>>(sum2, sq2, mean2, var2, 64, 1.f / (4096.f * 196.f));
    bn2_pool<<<50176, 256, 0, stream>>>(c2raw, mean2, var2, g2, b2, h2);
    fc1_wmma<<<256, 256, 0, stream>>>(fwh, h2, fb, feat);
    rls_kernel<<<1, 256, 0, stream>>>(feat, yv, W0, P0, sep, Pws, Wout);
    final_gemm<<<160, 256, 0, stream>>>(feat, Wout, (float*)d_out);
}
